// CharRNN_81733227643180
// MI455X (gfx1250) — matrix operand order repaired and checked by reading, not compiled, not run on hardware
//
#include <hip/hip_runtime.h>
#include <hip/hip_bf16.h>
#include <math.h>

typedef _Float16 v8h  __attribute__((ext_vector_type(8)));
typedef _Float16 v16h __attribute__((ext_vector_type(16)));
typedef float    v8f  __attribute__((ext_vector_type(8)));

#define T_STEPS 2048
#define BATCH   512
#define NCH     128
#define EMBD    128
#define HID     512

// ---------------------------------------------------------------------------
// Prep 1: fold embedding lookup + input projection into one 128x512 table
//   pe[c][j] = wax_b[j] + sum_e emb[c][e] * wax_w[j][e]
// ---------------------------------------------------------------------------
__global__ void prep_pe_kernel(const float* __restrict__ emb,
                               const float* __restrict__ wax_w,
                               const float* __restrict__ wax_b,
                               float* __restrict__ pe) {
    int idx = blockIdx.x * blockDim.x + threadIdx.x;   // 0 .. NCH*HID-1
    int c = idx / HID;
    int j = idx - c * HID;
    float acc = wax_b[j];
    #pragma unroll 8
    for (int e = 0; e < EMBD; ++e)
        acc = fmaf(emb[c * EMBD + e], wax_w[j * EMBD + e], acc);
    pe[idx] = acc;
}

// Prep 2: weight conversion f32 -> f16 (WMMA operand precision)
__global__ void cvt_f16_kernel(const float* __restrict__ src,
                               _Float16* __restrict__ dst, int n) {
    int idx = blockIdx.x * blockDim.x + threadIdx.x;
    if (idx < n) dst[idx] = (_Float16)src[idx];
}

// Branchless saturating tanh: 1 - 2/(e^{2x}+1).
// Uses v_exp_f32 + v_rcp_f32; exact limits: x->+inf => 1, x->-inf => -1.
__device__ __forceinline__ float fast_tanh(float x) {
    const float e = __expf(2.0f * x);
    return 1.0f - 2.0f * __builtin_amdgcn_rcpf(e + 1.0f);
}

// Build a 16x32 f16 A-fragment for row m = lane%16, half = lane/16
// from a row-major f16 row base (contiguous order: k = 16*half + s).
__device__ __forceinline__ v16h load_afrag(const _Float16* __restrict__ row,
                                           int k0, int hi) {
    return *(const v16h*)&row[k0 + hi * 16];
}

// ---------------------------------------------------------------------------
// Persistent RNN kernel. 32 blocks x 256 threads (8 wave32 waves).
// Block owns batch rows [b0, b0+16); the 16x512 f16 hidden slice lives in
// LDS for all 2048 steps (recurrence is batch-row-local -> no grid sync).
// ---------------------------------------------------------------------------
__launch_bounds__(256, 1)
__global__ void rnn_persistent_kernel(const int* __restrict__ seq,
                                      const float* __restrict__ pe,
                                      const _Float16* __restrict__ waa16,
                                      const _Float16* __restrict__ wya16,
                                      const float* __restrict__ wya_b,
                                      float* __restrict__ ys,   // [T][B][NCH]
                                      float* __restrict__ hT) { // [B][HID]
    __shared__ _Float16 hbuf[16 * HID];   // 16 KB hidden slice (f16)

    const int tid  = threadIdx.x;
    const int wave = tid >> 5;
    const int lane = tid & 31;
    const int nn   = lane & 15;   // C/D column, B column, A row
    const int hi   = lane >> 4;   // lane-half selector
    const int b0   = blockIdx.x * 16;

    // h0 = 0
    for (int i = tid; i < 16 * HID; i += 256) hbuf[i] = (_Float16)0.0f;
    __syncthreads();

    const float ybias = wya_b[wave * 16 + nn];
    const int   c0    = wave * 16;
    const _Float16* __restrict__ arow = &hbuf[nn * HID];  // A rows for WMMA

    for (int t = 0; t < T_STEPS; ++t) {
        // Opaque zero offset, renewed every time-step. Weight-load addresses
        // depend on it, so LICM cannot hoist ~640 VGPRs of loop-invariant
        // weight fragments out of the 2048-step loop (which caused scratch
        // spills in earlier builds). Unlike laundering the pointers through
        // the asm (which degraded loads to flat_load, ticking DScnt too),
        // an integer offset keeps the global address space -> global_load.
        int toff = 0;
        asm volatile("" : "+v"(toff));
        const _Float16* wrow = &waa16[(wave * 64 + nn) * HID + hi * 16 + toff];
        const _Float16* yrow = &wya16[(c0 + nn) * HID + hi * 16 + toff];

        // ---- C init for GEMM1: gather pe rows by this step's characters ----
        int chs[8];
        #pragma unroll
        for (int r = 0; r < 8; ++r)
            chs[r] = seq[t * BATCH + b0 + r + 8 * hi];   // row m = r + 8*hi

        v8f acc[4];
        #pragma unroll
        for (int i = 0; i < 4; ++i) {
            const int j0 = (wave * 4 + i) * 16;
            #pragma unroll
            for (int r = 0; r < 8; ++r)
                acc[i][r] = pe[chs[r] * HID + j0 + nn];
        }

        // ---- GEMM1: acc += h(16x512) @ waa^T ------------------------------
        // K processed in 4 rolled chunks of 4 slabs. Per chunk: batch-issue
        // 16 B-fragment + 4 A-fragment loads (clauses), then 16 WMMAs with
        // progressive s_wait countdown. No loop-carried operand registers.
        for (int kq = 0; kq < 4; ++kq) {
            v16h bfr[4][4];
            v16h afr[4];
            #pragma unroll
            for (int u = 0; u < 4; ++u) {
                const int k0 = (kq * 4 + u) * 32;
                #pragma unroll
                for (int i = 0; i < 4; ++i)
                    bfr[u][i] = *(const v16h*)&wrow[i * 16 * HID + k0];
                afr[u] = load_afrag(arow, k0, hi);
            }
            #pragma unroll
            for (int u = 0; u < 4; ++u) {
                #pragma unroll
                for (int i = 0; i < 4; ++i)
                    acc[i] = __builtin_amdgcn_wmma_f32_16x16x32_f16(
                        false, afr[u], false, bfr[u][i], (short)0, acc[i],
                        false, false);
            }
        }

        __syncthreads();   // all reads of old h complete before overwrite

        // ---- tanh (branchless), convert, write new hidden slice ----
        #pragma unroll
        for (int i = 0; i < 4; ++i) {
            const int j0 = (wave * 4 + i) * 16;
            #pragma unroll
            for (int r = 0; r < 8; ++r) {
                const float a = fast_tanh(acc[i][r]);
                hbuf[(r + 8 * hi) * HID + j0 + nn] = (_Float16)a;
            }
        }

        __syncthreads();   // new h visible to all waves

        // ---- GEMM2: y = h_new @ wya^T + bias, same chunked structure ------
        v8f yacc;
        #pragma unroll
        for (int r = 0; r < 8; ++r) yacc[r] = ybias;

        for (int kq = 0; kq < 4; ++kq) {
            v16h afr[4], bfr[4];
            #pragma unroll
            for (int u = 0; u < 4; ++u) {
                const int k0 = (kq * 4 + u) * 32;
                afr[u] = load_afrag(arow, k0, hi);
                bfr[u] = *(const v16h*)&yrow[k0];
            }
            #pragma unroll
            for (int u = 0; u < 4; ++u)
                yacc = __builtin_amdgcn_wmma_f32_16x16x32_f16(
                    false, afr[u], false, bfr[u], (short)0, yacc, false,
                    false);
        }

        float* yout = ys + (size_t)t * BATCH * NCH;
        #pragma unroll
        for (int r = 0; r < 8; ++r)
            yout[(b0 + r + 8 * hi) * NCH + c0 + nn] = yacc[r];
        // next step's hbuf writes are fenced by its first __syncthreads.
    }

    // final hidden state -> f32 output tail
    for (int i = tid; i < 16 * HID; i += 256) {
        const int m = i / HID;
        const int j = i - m * HID;
        hT[(b0 + m) * HID + j] = (float)hbuf[i];
    }
}

// ---------------------------------------------------------------------------
extern "C" void kernel_launch(void* const* d_in, const int* in_sizes, int n_in,
                              void* d_out, int out_size, void* d_ws, size_t ws_size,
                              hipStream_t stream) {
    (void)in_sizes; (void)n_in; (void)out_size; (void)ws_size;

    const int*   seq   = (const int*)  d_in[0];   // [T][B] int32
    const float* emb   = (const float*)d_in[1];   // [128][128]
    const float* wax_w = (const float*)d_in[2];   // [512][128]
    const float* wax_b = (const float*)d_in[3];   // [512]
    const float* waa_w = (const float*)d_in[4];   // [512][512]
    const float* wya_w = (const float*)d_in[5];   // [128][512]
    const float* wya_b = (const float*)d_in[6];   // [128]

    float* out = (float*)d_out;
    float* ys  = out;                                   // T*B*NCH floats
    float* hT  = out + (size_t)T_STEPS * BATCH * NCH;   // B*HID floats

    char* ws = (char*)d_ws;
    float*    pe    = (float*)ws;                                      // 256 KB
    _Float16* waa16 = (_Float16*)(ws + NCH * HID * 4);                 // 512 KB
    _Float16* wya16 = (_Float16*)(ws + NCH * HID * 4 + HID * HID * 2); // 128 KB

    prep_pe_kernel<<<(NCH * HID) / 256, 256, 0, stream>>>(emb, wax_w, wax_b, pe);
    cvt_f16_kernel<<<(HID * HID) / 256, 256, 0, stream>>>(waa_w, waa16, HID * HID);
    cvt_f16_kernel<<<(NCH * HID) / 256, 256, 0, stream>>>(wya_w, wya16, NCH * HID);

    rnn_persistent_kernel<<<BATCH / 16, 256, 0, stream>>>(
        seq, pe, waa16, wya16, wya_b, ys, hT);
}
